// Informer_34411277975918
// MI455X (gfx1250) — matrix operand order repaired and checked by reading, not compiled, not run on hardware
//
#include <hip/hip_runtime.h>
#include <cmath>
#include <cstdint>

// ---------------------------------------------------------------------------
// Informer forward for MI455X (gfx1250, wave32, WMMA).
// GEMMs run on v_wmma_f32_16x16x32_f16 (f16 A/B staged in LDS, f32 accum),
// 128x64 block tile, 8 waves, double-buffered LDS (one barrier per K-step).
// Attention / LayerNorm / measure run on wave32 VALU with LDS / shfl reductions.
//
// Assumed d_in flatten order = setup_inputs() dict insertion order:
//  0 encoder_x  1 decoder_x  2 in_w  3 in_b
//  enc layer i (i=0..2), base 4+12*i:
//    +0 ow +1 ob +2 n1g +3 n1b +4 n2g +5 n2b +6 w1 +7 b1 +8 w2 +9 b2 +10 cw +11 cb
//  40 enc_ng 41 enc_nb
//  dec layer i (i=0..2), base 42+16*i:
//    +0 sow +1 sob +2 ciw +3 cib +4 cow +5 cob +6 n1g +7 n1b +8 n2g +9 n2b
//    +10 n3g +11 n3b +12 w1 +13 b1 +14 w2 +15 b2
//  90 dec_ng 91 dec_nb 92 out_w 93 out_b
// ---------------------------------------------------------------------------

#define BATCH   8
#define DMODEL  512
#define NHEAD   8
#define HD      64
#define DFF     2048
#define INDIM   32
#define ENCLEN  1024
#define DECLEN  768          // DEC_LEN + PRED_LEN
#define PREDLEN 256

typedef __attribute__((ext_vector_type(16))) _Float16 v16h;
typedef __attribute__((ext_vector_type(8)))  float    v8f;
typedef __attribute__((ext_vector_type(8)))  _Float16 h8;

__device__ __forceinline__ h8 inf_cvt8(float4 a, float4 b) {
  h8 r;
  r[0] = (_Float16)a.x; r[1] = (_Float16)a.y; r[2] = (_Float16)a.z; r[3] = (_Float16)a.w;
  r[4] = (_Float16)b.x; r[5] = (_Float16)b.y; r[6] = (_Float16)b.z; r[7] = (_Float16)b.w;
  return r;
}

// Stage a 128x32 f32 tile -> f16 LDS [row][k] (256 threads, 16 elems each).
// Requires ldk % 32 == 0 and 16B-aligned src rows (true at all call sites).
__device__ __forceinline__ void inf_stageA(const float* __restrict__ src, int row0,
                                           int rv, int ldk, int k0,
                                           _Float16 (*dst)[40], int tid) {
  const int r = tid >> 1, seg = (tid & 1) << 4;
  h8 lo, hi;
  if (r < rv) {
    const float4* p = (const float4*)(src + (size_t)(row0 + r) * ldk + k0 + seg);
    float4 a = p[0], b = p[1], c = p[2], d = p[3];
    lo = inf_cvt8(a, b);
    hi = inf_cvt8(c, d);
  } else {
#pragma unroll
    for (int i = 0; i < 8; ++i) { lo[i] = (_Float16)0.f; hi[i] = (_Float16)0.f; }
  }
  *(h8*)&dst[r][seg]     = lo;
  *(h8*)&dst[r][seg + 8] = hi;
}

// Stage a 64x32 f32 tile -> f16 LDS [row][k] (256 threads, 8 elems each).
__device__ __forceinline__ void inf_stageB(const float* __restrict__ src, int row0,
                                           int rv, int ldk, int k0,
                                           _Float16 (*dst)[40], int tid) {
  const int r = tid >> 2, seg = (tid & 3) << 3;
  h8 v;
  if (r < rv) {
    const float4* p = (const float4*)(src + (size_t)(row0 + r) * ldk + k0 + seg);
    v = inf_cvt8(p[0], p[1]);
  } else {
#pragma unroll
    for (int i = 0; i < 8; ++i) v[i] = (_Float16)0.f;
  }
  *(h8*)&dst[r][seg] = v;
}

// ============================ WMMA GEMM ====================================
// C[M,N] = act( A[M,K] @ W[N,K]^T + bias ), act: 0=none, 1=GELU(erf), 2=ELU
// Preconditions (guaranteed by the driver): M % 128 == 0, K % 32 == 0.
// N is arbitrary (ragged rows zero-filled in staging, stores guarded).
// Block: 128x64 tile, 256 threads = 8 waves, each wave 32x32 (2x2 WMMA).
__global__ __launch_bounds__(256)
void inf_gemm(const float* __restrict__ A, const float* __restrict__ W,
              const float* __restrict__ bias, float* __restrict__ C,
              int M, int N, int K, int act) {
  __shared__ _Float16 As[2][128][40];   // [m][k], +8 pad
  __shared__ _Float16 Bs[2][64][40];    // [n][k], +8 pad (same layout as A!)

  const int tid  = threadIdx.x;
  const int wv   = tid >> 5;
  const int lane = tid & 31;
  const int wm   = wv >> 1, wn = wv & 1;   // 4x2 wave grid
  const int half = lane >> 4;              // lane group 0/1
  const int l16  = lane & 15;
  const int m0 = blockIdx.y * 128, n0 = blockIdx.x * 64;

  int av = M - m0; if (av > 128) av = 128;   // always 128 at our call sites
  int bv = N - n0; if (bv > 64)  bv = 64;    // ragged only for the output head

  inf_stageA(A, m0, av, K, 0, As[0], tid);
  inf_stageB(W, n0, bv, K, 0, Bs[0], tid);

  const int nk = K >> 5;
  v8f c[2][2] = {};
  int cur = 0;

  for (int t = 0; t < nk; ++t) {
    __syncthreads();
    if (t + 1 < nk) {               // overlap next-tile staging with WMMA
      inf_stageA(A, m0, av, K, (t + 1) << 5, As[cur ^ 1], tid);
      inf_stageB(W, n0, bv, K, (t + 1) << 5, Bs[cur ^ 1], tid);
    }
    if (t + 2 < nk) {               // L2 prefetch (global_prefetch_b8)
      int r = tid >> 1;
      if (r < av)
        __builtin_prefetch(&A[(size_t)(m0 + r) * K + ((t + 2) << 5) + ((tid & 1) << 4)], 0, 1);
    }

    // Fragments per ISA 7.12.2. Both gathers are two runs of 8 contiguous halves per
    // lane -> 2x ds_load_b128 each.
    v16h af[2], bf[2];
#pragma unroll
    for (int tm = 0; tm < 2; ++tm)
#pragma unroll
      for (int i = 0; i < 16; ++i) {
        int kk = ((i >> 3) << 4) + (half << 3) + (i & 7);   // A: K groups of 8
        af[tm][i] = As[cur][wm * 32 + tm * 16 + l16][kk];
      }
#pragma unroll
    for (int tn = 0; tn < 2; ++tn)
#pragma unroll
      for (int i = 0; i < 16; ++i)                           // B: K groups of 8
        bf[tn][i] = Bs[cur][wn * 32 + tn * 16 + l16][((i >> 3) << 4) + (half << 3) + (i & 7)];

#pragma unroll
    for (int tm = 0; tm < 2; ++tm)
#pragma unroll
      for (int tn = 0; tn < 2; ++tn)
        c[tm][tn] = __builtin_amdgcn_wmma_f32_16x16x32_f16(
            false, af[tm], false, bf[tn], (short)0, c[tm][tn], false, false);
    cur ^= 1;
  }

  // ---- epilogue: bias + activation, C/D layout (M = half*8 + r, N = l16)
#pragma unroll
  for (int tm = 0; tm < 2; ++tm)
#pragma unroll
    for (int tn = 0; tn < 2; ++tn)
#pragma unroll
      for (int r = 0; r < 8; ++r) {
        int m = m0 + wm * 32 + tm * 16 + half * 8 + r;   // m < M by precondition
        int n = n0 + wn * 32 + tn * 16 + l16;
        if (n < N) {
          float v = c[tm][tn][r] + (bias ? bias[n] : 0.f);
          if (act == 1)       v = 0.5f * v * (1.f + erff(v * 0.70710678118654752f));
          else if (act == 2)  v = (v > 0.f) ? v : (__expf(v) - 1.f);
          C[(size_t)m * N + n] = v;
        }
      }
}

// ====================== ProbSparse: measure (m1 - m2) ======================
// grid (B*H, L/8), block 256 (8 waves: wave w -> query row i0+w).
__global__ __launch_bounds__(256)
void inf_measure(const float* __restrict__ x, float* __restrict__ meas, int L) {
  const int bh = blockIdx.x, b = bh >> 3, h = bh & 7;
  const int i0 = blockIdx.y * 8;
  const int tid = threadIdx.x, w = tid >> 5, lane = tid & 31;
  __shared__ float Qs[8][HD];
  __shared__ float Ks[64][HD + 4];
  const float* xb = x + (size_t)b * L * DMODEL + h * HD;

  for (int t = tid; t < 8 * HD; t += 256) {
    int r = t >> 6, d = t & 63;
    Qs[r][d] = xb[(size_t)(i0 + r) * DMODEL + d];
  }
  float m = -INFINITY, s = 0.f, msum = 0.f;
  const float scale = 0.125f;   // 1/sqrt(64)

  for (int kt = 0; kt < L; kt += 64) {
    __syncthreads();
    for (int t = tid; t < 64 * HD; t += 256) {
      int r = t >> 6, d = t & 63;
      Ks[r][d] = xb[(size_t)(kt + r) * DMODEL + d];
    }
    __syncthreads();
#pragma unroll
    for (int jj = 0; jj < 64; jj += 32) {
      int j = jj + lane;
      float acc = 0.f;
#pragma unroll 8
      for (int d = 0; d < HD; ++d) acc += Qs[w][d] * Ks[j][d];
      acc *= scale;
      msum += acc;
      float nm = fmaxf(m, acc);
      s = s * __expf(m - nm) + __expf(acc - nm);
      m = nm;
    }
  }
  // merge online-LSE state across the wave
  for (int off = 16; off > 0; off >>= 1) {
    float om = __shfl_xor(m, off, 32);
    float os = __shfl_xor(s, off, 32);
    float osum = __shfl_xor(msum, off, 32);
    float nm = fmaxf(m, om);
    s = s * __expf(m - nm) + os * __expf(om - nm);
    m = nm;
    msum += osum;
  }
  if (lane == 0) {
    float m1 = m + __logf(s) - __logf((float)L);
    float m2 = msum / (float)L;
    meas[(size_t)bh * L + (i0 + w)] = m1 - m2;
  }
}

// ===================== top-u by repeated wave argmax ========================
__global__ void inf_topk(const float* __restrict__ meas, int* __restrict__ idx,
                         int L, int u) {
  const int bh = blockIdx.x, lane = threadIdx.x;   // block = 32
  const float* mr = meas + (size_t)bh * L;
  int sel[8];
  for (int p = 0; p < u; ++p) {
    float bv = -INFINITY; int bj = 0x7fffffff;
    for (int j = lane; j < L; j += 32) {
      bool used = false;
      for (int q = 0; q < p; ++q) used = used || (sel[q] == j);
      if (used) continue;
      float v = mr[j];
      if (v > bv || (v == bv && j < bj)) { bv = v; bj = j; }
    }
    for (int off = 16; off > 0; off >>= 1) {
      float ov = __shfl_xor(bv, off, 32);
      int   oj = __shfl_xor(bj, off, 32);
      if (ov > bv || (ov == bv && oj < bj)) { bv = ov; bj = oj; }
    }
    sel[p] = bj;
    if (lane == 0) idx[bh * u + p] = bj;
  }
}

// ======================= v_avg = mean_l q[b,h,l,:] ==========================
__global__ void inf_vavg(const float* __restrict__ x, float* __restrict__ vavg, int L) {
  const int bh = blockIdx.x, b = bh >> 3, h = bh & 7, d = threadIdx.x; // 64 thr
  const float* xb = x + (size_t)b * L * DMODEL + h * HD + d;
  float s = 0.f;
  for (int l = 0; l < L; ++l) s += xb[(size_t)l * DMODEL];
  vavg[bh * HD + d] = s / (float)L;
}

// =============== broadcast v_avg to every row (recombined layout) ==========
__global__ void inf_bcast(const float* __restrict__ vavg, float* __restrict__ attn,
                          int L) {
  size_t n = (size_t)BATCH * L * DMODEL;
  size_t t = (size_t)blockIdx.x * 256 + threadIdx.x;
  if (t >= n) return;
  int c = (int)(t % DMODEL);
  int b = (int)(t / ((size_t)L * DMODEL));
  attn[t] = vavg[(b * NHEAD + (c >> 6)) * HD + (c & 63)];
}

// ============ active-query softmax attention (overwrites rows) =============
__global__ __launch_bounds__(256)
void inf_sparse_attn(const float* __restrict__ x, const int* __restrict__ idx,
                     float* __restrict__ attn, int L, int u, int causal) {
  const int g = blockIdx.x, uu = g % u, bh = g / u, b = bh >> 3, h = bh & 7;
  const int i = idx[bh * u + uu];
  const int tid = threadIdx.x;
  __shared__ float q[HD];
  __shared__ float sc[1024];
  __shared__ float red[256];
  const float* xb = x + (size_t)b * L * DMODEL + h * HD;

  if (tid < HD) q[tid] = xb[(size_t)i * DMODEL + tid];
  __syncthreads();

  for (int j = tid; j < L; j += 256) {
    float acc = 0.f;
    const float* kj = xb + (size_t)j * DMODEL;
#pragma unroll 8
    for (int d = 0; d < HD; ++d) acc += q[d] * kj[d];
    acc *= 0.125f;
    if (causal && j > i) acc = -INFINITY;
    sc[j] = acc;
  }
  __syncthreads();
  float lm = -INFINITY;
  for (int j = tid; j < L; j += 256) lm = fmaxf(lm, sc[j]);
  red[tid] = lm; __syncthreads();
  for (int o = 128; o > 0; o >>= 1) { if (tid < o) red[tid] = fmaxf(red[tid], red[tid + o]); __syncthreads(); }
  float mx = red[0]; __syncthreads();
  float ls = 0.f;
  for (int j = tid; j < L; j += 256) { float e = __expf(sc[j] - mx); sc[j] = e; ls += e; }
  red[tid] = ls; __syncthreads();
  for (int o = 128; o > 0; o >>= 1) { if (tid < o) red[tid] += red[tid + o]; __syncthreads(); }
  float inv = 1.f / red[0]; __syncthreads();

  const int d = tid & 63, gg = tid >> 6;
  float acc = 0.f;
  for (int j = gg; j < L; j += 4) acc += sc[j] * xb[(size_t)j * DMODEL + d];
  red[tid] = acc; __syncthreads();
  if (tid < HD) {
    float o = (red[tid] + red[64 + tid] + red[128 + tid] + red[192 + tid]) * inv;
    attn[((size_t)b * L + i) * DMODEL + h * HD + tid] = o;
  }
}

// ======================== full cross attention ==============================
// grid = B*H*Lq, block 256; Lk <= 256.
__global__ __launch_bounds__(256)
void inf_cross_attn(const float* __restrict__ q, const float* __restrict__ kk,
                    const float* __restrict__ vv, float* __restrict__ out,
                    int Lq, int Lk) {
  const int g = blockIdx.x, i = g % Lq, bh = g / Lq, b = bh >> 3, h = bh & 7;
  const int tid = threadIdx.x;
  __shared__ float qs[HD];
  __shared__ float sc[256];
  __shared__ float red[256];
  const float* qr = q  + ((size_t)b * Lq + i) * DMODEL + h * HD;
  const float* kb = kk + (size_t)b * Lk * DMODEL + h * HD;
  const float* vb = vv + (size_t)b * Lk * DMODEL + h * HD;

  if (tid < HD) qs[tid] = qr[tid];
  __syncthreads();
  float lm = -INFINITY;
  for (int j = tid; j < Lk; j += 256) {
    float acc = 0.f;
    const float* kj = kb + (size_t)j * DMODEL;
#pragma unroll 8
    for (int d = 0; d < HD; ++d) acc += qs[d] * kj[d];
    acc *= 0.125f;
    sc[j] = acc;
    lm = fmaxf(lm, acc);
  }
  red[tid] = lm; __syncthreads();
  for (int o = 128; o > 0; o >>= 1) { if (tid < o) red[tid] = fmaxf(red[tid], red[tid + o]); __syncthreads(); }
  float mx = red[0]; __syncthreads();
  float ls = 0.f;
  for (int j = tid; j < Lk; j += 256) { float e = __expf(sc[j] - mx); sc[j] = e; ls += e; }
  red[tid] = ls; __syncthreads();
  for (int o = 128; o > 0; o >>= 1) { if (tid < o) red[tid] += red[tid + o]; __syncthreads(); }
  float inv = 1.f / red[0]; __syncthreads();

  const int d = tid & 63, gg = tid >> 6;
  float acc = 0.f;
  for (int j = gg; j < Lk; j += 4) acc += sc[j] * vb[(size_t)j * DMODEL + d];
  red[tid] = acc; __syncthreads();
  if (tid < HD) {
    float o = (red[tid] + red[64 + tid] + red[128 + tid] + red[192 + tid]) * inv;
    out[((size_t)b * Lq + i) * DMODEL + h * HD + tid] = o;
  }
}

// ===================== residual add + LayerNorm (D=512) =====================
__global__ __launch_bounds__(256)
void inf_add_ln(const float* __restrict__ x, const float* __restrict__ a,
                const float* __restrict__ gam, const float* __restrict__ bet,
                float* __restrict__ out) {
  const size_t row = blockIdx.x;
  const int tid = threadIdx.x;
  __shared__ float red[256];
  const float* xr = x + row * DMODEL;
  float v0 = xr[tid]       + (a ? a[row * DMODEL + tid]       : 0.f);
  float v1 = xr[tid + 256] + (a ? a[row * DMODEL + tid + 256] : 0.f);
  red[tid] = v0 + v1; __syncthreads();
  for (int o = 128; o > 0; o >>= 1) { if (tid < o) red[tid] += red[tid + o]; __syncthreads(); }
  float mean = red[0] * (1.f / DMODEL); __syncthreads();
  float d0 = v0 - mean, d1 = v1 - mean;
  red[tid] = d0 * d0 + d1 * d1; __syncthreads();
  for (int o = 128; o > 0; o >>= 1) { if (tid < o) red[tid] += red[tid + o]; __syncthreads(); }
  float rstd = rsqrtf(red[0] * (1.f / DMODEL) + 1e-5f);
  out[row * DMODEL + tid]       = d0 * rstd * gam[tid]       + bet[tid];
  out[row * DMODEL + tid + 256] = d1 * rstd * gam[tid + 256] + bet[tid + 256];
}

// ====================== misc elementwise kernels ============================
__global__ void inf_add_pe(float* __restrict__ x, int L) {
  size_t n = (size_t)BATCH * L * DMODEL;
  size_t t = (size_t)blockIdx.x * 256 + threadIdx.x;
  if (t >= n) return;
  int c = (int)(t % DMODEL);
  int l = (int)((t / DMODEL) % (size_t)L);
  int p = c >> 1;
  float freq = __expf((float)(2 * p) * (-9.210340371976184f / (float)DMODEL));
  float ang = (float)l * freq;
  x[t] += (c & 1) ? cosf(ang) : sinf(ang);
}

__global__ void inf_build_decin(const float* __restrict__ dx, float* __restrict__ out) {
  int t = blockIdx.x * 256 + threadIdx.x;
  if (t >= BATCH * DECLEN * INDIM) return;
  int c = t % INDIM, r = (t / INDIM) % DECLEN, b = t / (INDIM * DECLEN);
  out[t] = (r < 512) ? dx[(b * 512 + r) * INDIM + c] : 0.f;
}

__global__ void inf_im2col(const float* __restrict__ x, float* __restrict__ out, int L) {
  size_t n = (size_t)BATCH * L * (3 * DMODEL);
  size_t t = (size_t)blockIdx.x * 256 + threadIdx.x;
  if (t >= n) return;
  int c = (int)(t % (3 * DMODEL));
  size_t bl = t / (3 * DMODEL);
  int l = (int)(bl % (size_t)L), b = (int)(bl / (size_t)L);
  int tap = c / DMODEL, i = c % DMODEL;
  int ls = l + tap - 1;
  out[t] = (ls >= 0 && ls < L) ? x[((size_t)b * L + ls) * DMODEL + i] : 0.f;
}

__global__ void inf_repack_cw(const float* __restrict__ cw, float* __restrict__ wc) {
  int t = blockIdx.x * 256 + threadIdx.x;
  if (t >= DMODEL * 3 * DMODEL) return;
  int o = t / (3 * DMODEL), c = t % (3 * DMODEL);
  int tap = c / DMODEL, i = c % DMODEL;
  wc[t] = cw[(size_t)(o * DMODEL + i) * 3 + tap];
}

__global__ void inf_maxpool(const float* __restrict__ y, float* __restrict__ out, int Lh) {
  size_t n = (size_t)BATCH * Lh * DMODEL;
  size_t t = (size_t)blockIdx.x * 256 + threadIdx.x;
  if (t >= n) return;
  int d = (int)(t % DMODEL);
  size_t bl = t / DMODEL;
  int l = (int)(bl % (size_t)Lh), b = (int)(bl / (size_t)Lh);
  const float* base = y + (((size_t)b * (2 * Lh) + 2 * l) * DMODEL) + d;
  out[t] = fmaxf(base[0], base[DMODEL]);
}

__global__ void inf_slice_out(const float* __restrict__ full, float* __restrict__ out) {
  int t = blockIdx.x * 256 + threadIdx.x;
  if (t >= BATCH * PREDLEN * INDIM) return;
  int c = t % INDIM, r = (t / INDIM) % PREDLEN, b = t / (INDIM * PREDLEN);
  out[t] = full[((size_t)b * DECLEN + (512 + r)) * INDIM + c];
}

// ============================== driver ======================================
extern "C" void kernel_launch(void* const* d_in, const int* in_sizes, int n_in,
                              void* d_out, int out_size, void* d_ws, size_t ws_size,
                              hipStream_t stream) {
  (void)in_sizes; (void)n_in; (void)out_size; (void)ws_size;
  auto F = [&](int i) { return (const float*)d_in[i]; };

  const float* encoder_x = F(0);
  const float* decoder_x = F(1);
  const float* in_w = F(2);
  const float* in_b = F(3);

  // workspace arena (fp32)
  float* ws = (float*)d_ws;
  size_t off = 0;
  auto alloc = [&](size_t nf) { float* p = ws + off; off += (nf + 63) & ~(size_t)63; return p; };
  float* X     = alloc((size_t)8192 * DMODEL);      // activations
  float* Ab    = alloc((size_t)8192 * DMODEL);      // attn / ffn out
  float* T     = alloc((size_t)8192 * DMODEL);      // temp (attn rows / conv out)
  float* H     = alloc((size_t)8192 * DFF);         // ffn hidden / im2col / cross-Q
  float* MEMB  = alloc((size_t)BATCH * 128 * DMODEL);
  float* KC    = alloc((size_t)BATCH * 128 * DMODEL);
  float* VC    = alloc((size_t)BATCH * 128 * DMODEL);
  float* MEAS  = alloc((size_t)BATCH * NHEAD * 1024);
  int*   IDX   = (int*)alloc(512);
  float* VAVG  = alloc((size_t)BATCH * NHEAD * HD);
  float* WC    = alloc((size_t)DMODEL * 3 * DMODEL);
  float* DECIN = alloc((size_t)BATCH * DECLEN * INDIM);
  float* OUTT  = alloc((size_t)BATCH * DECLEN * INDIM);

  // M % 128 == 0 and K % 32 == 0 hold for every call below.
  auto gemm = [&](const float* A, const float* W, const float* b, float* C,
                  int M, int N, int K, int act) {
    dim3 g((N + 63) / 64, (M + 127) / 128);
    inf_gemm<<<g, 256, 0, stream>>>(A, W, b, C, M, N, K, act);
  };
  auto addln = [&](const float* x, const float* a, const float* g, const float* b,
                   float* out, int rows) {
    inf_add_ln<<<rows, 256, 0, stream>>>(x, a, g, b, out);
  };
  auto probsparse = [&](float* x, const float* Wo, const float* bo, float* aout,
                        int L, int causal) {
    int u = (int)std::ceil(std::log((double)L + 1.0));
    if (u < 1) u = 1;
    inf_measure<<<dim3(BATCH * NHEAD, L / 8), 256, 0, stream>>>(x, MEAS, L);
    inf_topk<<<BATCH * NHEAD, 32, 0, stream>>>(MEAS, IDX, L, u);
    inf_vavg<<<BATCH * NHEAD, HD, 0, stream>>>(x, VAVG, L);
    size_t n = (size_t)BATCH * L * DMODEL;
    inf_bcast<<<(unsigned)((n + 255) / 256), 256, 0, stream>>>(VAVG, T, L);
    inf_sparse_attn<<<BATCH * NHEAD * u, 256, 0, stream>>>(x, IDX, T, L, u, causal);
    gemm(T, Wo, bo, aout, BATCH * L, DMODEL, DMODEL, 0);
  };

  // ------------------------------ encoder ------------------------------
  gemm(encoder_x, in_w, in_b, X, BATCH * ENCLEN, DMODEL, INDIM, 0);
  {
    size_t n = (size_t)BATCH * ENCLEN * DMODEL;
    inf_add_pe<<<(unsigned)((n + 255) / 256), 256, 0, stream>>>(X, ENCLEN);
  }
  int L = ENCLEN;
  for (int l = 0; l < 3; ++l) {
    int base = 4 + l * 12;
    probsparse(X, F(base + 0), F(base + 1), Ab, L, 0);
    addln(X, Ab, F(base + 2), F(base + 3), X, BATCH * L);
    gemm(X, F(base + 6), F(base + 7), H, BATCH * L, DFF, DMODEL, 1);   // GELU
    gemm(H, F(base + 8), F(base + 9), Ab, BATCH * L, DMODEL, DFF, 0);
    addln(X, Ab, F(base + 4), F(base + 5), X, BATCH * L);
    // distill: conv1d(k=3,pad=1) as im2col GEMM + ELU + maxpool(2)
    inf_repack_cw<<<(DMODEL * 3 * DMODEL + 255) / 256, 256, 0, stream>>>(F(base + 10), WC);
    {
      size_t n = (size_t)BATCH * L * 3 * DMODEL;
      inf_im2col<<<(unsigned)((n + 255) / 256), 256, 0, stream>>>(X, H, L);
    }
    gemm(H, WC, F(base + 11), T, BATCH * L, DMODEL, 3 * DMODEL, 2);    // ELU
    {
      size_t n = (size_t)BATCH * (L / 2) * DMODEL;
      inf_maxpool<<<(unsigned)((n + 255) / 256), 256, 0, stream>>>(T, X, L / 2);
    }
    L /= 2;
  }
  addln(X, nullptr, F(40), F(41), MEMB, BATCH * L);   // memory, L == 128
  const int LMEM = L;

  // ------------------------------ decoder ------------------------------
  inf_build_decin<<<(BATCH * DECLEN * INDIM + 255) / 256, 256, 0, stream>>>(decoder_x, DECIN);
  gemm(DECIN, in_w, in_b, X, BATCH * DECLEN, DMODEL, INDIM, 0);
  {
    size_t n = (size_t)BATCH * DECLEN * DMODEL;
    inf_add_pe<<<(unsigned)((n + 255) / 256), 256, 0, stream>>>(X, DECLEN);
  }
  for (int l = 0; l < 3; ++l) {
    int base = 42 + l * 16;
    probsparse(X, F(base + 0), F(base + 1), Ab, DECLEN, 1);            // causal
    addln(X, Ab, F(base + 6), F(base + 7), X, BATCH * DECLEN);
    // cross attention: q from y, k/v from memory (ciw split rows)
    const float* ciw = F(base + 2);
    const float* cib = F(base + 3);
    gemm(X,    ciw,                               cib,              H,  BATCH * DECLEN, DMODEL, DMODEL, 0);
    gemm(MEMB, ciw + (size_t)DMODEL * DMODEL,     cib + DMODEL,     KC, BATCH * LMEM, DMODEL, DMODEL, 0);
    gemm(MEMB, ciw + (size_t)2 * DMODEL * DMODEL, cib + 2 * DMODEL, VC, BATCH * LMEM, DMODEL, DMODEL, 0);
    inf_cross_attn<<<BATCH * NHEAD * DECLEN, 256, 0, stream>>>(H, KC, VC, T, DECLEN, LMEM);
    gemm(T, F(base + 4), F(base + 5), Ab, BATCH * DECLEN, DMODEL, DMODEL, 0);
    addln(X, Ab, F(base + 8), F(base + 9), X, BATCH * DECLEN);
    gemm(X, F(base + 12), F(base + 13), H, BATCH * DECLEN, DFF, DMODEL, 1);  // GELU
    gemm(H, F(base + 14), F(base + 15), Ab, BATCH * DECLEN, DMODEL, DFF, 0);
    addln(X, Ab, F(base + 10), F(base + 11), X, BATCH * DECLEN);
  }
  addln(X, nullptr, F(90), F(91), T, BATCH * DECLEN);
  gemm(T, F(92), F(93), OUTT, BATCH * DECLEN, INDIM, DMODEL, 0);
  inf_slice_out<<<(BATCH * PREDLEN * INDIM + 255) / 256, 256, 0, stream>>>(OUTT, (float*)d_out);
}
